// MyMultiHeadAttention_40750649704982
// MI455X (gfx1250) — matrix operand order repaired and checked by reading, not compiled, not run on hardware
//
#include <hip/hip_runtime.h>
#include <hip/hip_bf16.h>

// ---------------------------------------------------------------------------
// Multi-head attention for MI455X (gfx1250), wave32 + WMMA f32_16x16x32_f16.
//   L=4096, D=1024, H=16, Dh=64
// Pipeline:
//   1) fp32 -> f16 conversion of Q,K,V and w_q,w_k,w_v,w_o into workspace
//   2) GEMM  q = Qh*Wq^T + b  (f16 out, row-major)
//            k = Kh*Wk^T + b  (f16 out, row-major)
//            v = Vh*Wv^T + b  (f16 out, TRANSPOSED: vT[n*L + m])
//   3) flash attention per (head, 16-query-row block), streaming softmax,
//      KV step 64, DPP reductions, exp2-domain softmax, prefetch next block
//   4) GEMM  out = attn*Wo^T + b_o (f32 out, row-major -> d_out)
// GEMM staging: GLOBAL_LOAD_ASYNC_TO_LDS_B128 (inline asm, ASYNCcnt-tracked)
// double-buffered in LDS.
// ---------------------------------------------------------------------------

typedef _Float16 v16h __attribute__((ext_vector_type(16)));
typedef _Float16 v8h  __attribute__((ext_vector_type(8)));
typedef float    v8f  __attribute__((ext_vector_type(8)));

#define L_SEQ 4096
#define D_MODEL 1024
#define N_HEADS 16
#define D_HEAD 64

// ---- async global->LDS copy (ISA 15.18 VGLOBAL, GV mode) ------------------
// vdst = VGPR holding LDS byte offset; vaddr = 64-bit global address.
// Generic shared pointers carry the LDS offset in their low 32 bits
// (addrspacecast AS3->generic builds {SHARED_BASE_hi, lds_offset}).
static __device__ __forceinline__ void async_load_b128(const void* g,
                                                       void* lds) {
  unsigned int loff = (unsigned int)(unsigned long long)lds;
  asm volatile("global_load_async_to_lds_b128 %0, %1, off"
               :
               : "v"(loff), "v"(g)
               : "memory");
}

static __device__ __forceinline__ void wait_async_le3() {
#if __has_builtin(__builtin_amdgcn_s_wait_asynccnt)
  __builtin_amdgcn_s_wait_asynccnt(3);
#else
  asm volatile("s_wait_asynccnt 0x3" ::: "memory");
#endif
}
static __device__ __forceinline__ void wait_async_le0() {
#if __has_builtin(__builtin_amdgcn_s_wait_asynccnt)
  __builtin_amdgcn_s_wait_asynccnt(0);
#else
  asm volatile("s_wait_asynccnt 0x0" ::: "memory");
#endif
}

static __device__ __forceinline__ v8f wmma_f16(v16h a, v16h b, v8f c) {
  // D = A(16x32 f16) * B(32x16 f16) + C(16x16 f32)
  return __builtin_amdgcn_wmma_f32_16x16x32_f16(
      /*neg_a=*/false, a, /*neg_b=*/false, b,
      /*c_mod=*/(short)0, c, /*reuse_a=*/false, /*reuse_b=*/false);
}

static __device__ __forceinline__ v16h cat8(v8h lo, v8h hi) {
  return __builtin_shufflevector(lo, hi, 0, 1, 2, 3, 4, 5, 6, 7,
                                 8, 9, 10, 11, 12, 13, 14, 15);
}

// --------------------- 16-lane xor reductions (DPP) -----------------------
// quad_perm{1,0,3,2}=0xB1 (xor1), quad_perm{2,3,0,1}=0x4E (xor2),
// row_half_mirror=0x141 (xor4-equivalent once quads uniform),
// row_mirror=0x140 (xor8-equivalent once 8-groups uniform).
#if __has_builtin(__builtin_amdgcn_update_dpp)
template <int CTRL>
static __device__ __forceinline__ float dppf(float x) {
  int i = __float_as_int(x);
  return __int_as_float(__builtin_amdgcn_update_dpp(i, i, CTRL, 0xF, 0xF, true));
}
static __device__ __forceinline__ float redmax16(float v) {
  v = fmaxf(v, dppf<0xB1>(v));
  v = fmaxf(v, dppf<0x4E>(v));
  v = fmaxf(v, dppf<0x141>(v));
  v = fmaxf(v, dppf<0x140>(v));
  return v;
}
static __device__ __forceinline__ float redsum16(float v) {
  v += dppf<0xB1>(v);
  v += dppf<0x4E>(v);
  v += dppf<0x141>(v);
  v += dppf<0x140>(v);
  return v;
}
#else
static __device__ __forceinline__ float redmax16(float v) {
  v = fmaxf(v, __shfl_xor(v, 1, 32));
  v = fmaxf(v, __shfl_xor(v, 2, 32));
  v = fmaxf(v, __shfl_xor(v, 4, 32));
  v = fmaxf(v, __shfl_xor(v, 8, 32));
  return v;
}
static __device__ __forceinline__ float redsum16(float v) {
  v += __shfl_xor(v, 1, 32);
  v += __shfl_xor(v, 2, 32);
  v += __shfl_xor(v, 4, 32);
  v += __shfl_xor(v, 8, 32);
  return v;
}
#endif

// ---------------------------------------------------------------------------
// fp32 -> f16 conversion (4 elements / thread)
// ---------------------------------------------------------------------------
__global__ void cvt_f32_f16(const float* __restrict__ in,
                            _Float16* __restrict__ out, int n4) {
  int i = blockIdx.x * blockDim.x + threadIdx.x;
  if (i < n4) {
    float4 v = ((const float4*)in)[i];
    _Float16* o = out + (size_t)i * 4;
    o[0] = (_Float16)v.x;
    o[1] = (_Float16)v.y;
    o[2] = (_Float16)v.z;
    o[3] = (_Float16)v.w;
  }
}

// ---------------------------------------------------------------------------
// GEMM: C[M,N] = A[M,K](f16) * W[N,K]^T(f16) + bias[N](f32)
// MODE 0: f16 out row-major | MODE 1: f16 out transposed | MODE 2: f32 out
// WG: 256 threads = 8 waves; tile 128(M) x 64(N); K-step 32, double-buffered
// in LDS via async global->LDS b128 (each thread: 3 ops of 16B per stage).
// ---------------------------------------------------------------------------
template <int MODE>
__global__ __launch_bounds__(256) void gemm_bt(
    const _Float16* __restrict__ A, const _Float16* __restrict__ W,
    const float* __restrict__ bias, void* __restrict__ outp,
    int M, int N, int K) {
  const int tid = threadIdx.x;
  const int lane = tid & 31;
  const int wave = tid >> 5;
  const int mw = wave & 3;
  const int nw = wave >> 2;
  const int lo16 = lane & 15;
  const int hi = lane >> 4;
  const int mBase = blockIdx.y * 128;
  const int nBase = blockIdx.x * 64;

  v8f c[2][2] = {};

  __shared__ __align__(64) _Float16 As[2][128 * 32];
  __shared__ __align__(64) _Float16 Bs[2][64 * 32];

  // Each thread issues 3 async b128 (16B per lane) per K-step:
  //   2 chunks of the 128x32 A tile, 1 chunk of the 64x32 W tile.
  auto stage = [&](int buf, int k0) {
    {
      int c0 = tid, row = c0 >> 2, part = c0 & 3;
      async_load_b128(A + (size_t)(mBase + row) * K + k0 + part * 8,
                      &As[buf][row * 32 + part * 8]);
    }
    {
      int c1 = tid + 256, row = c1 >> 2, part = c1 & 3;
      async_load_b128(A + (size_t)(mBase + row) * K + k0 + part * 8,
                      &As[buf][row * 32 + part * 8]);
    }
    {
      int row = tid >> 2, part = tid & 3;
      async_load_b128(W + (size_t)(nBase + row) * K + k0 + part * 8,
                      &Bs[buf][row * 32 + part * 8]);
    }
  };

  stage(0, 0);
  for (int k0 = 0; k0 < K; k0 += 32) {
    int buf = (k0 >> 5) & 1;
    if (k0 + 32 < K) {
      stage(buf ^ 1, k0 + 32);
      wait_async_le3();  // current buffer's 3 ops (issued first) are done
    } else {
      wait_async_le0();
    }
    __syncthreads();

    v16h a[2], b[2];
#pragma unroll
    for (int i = 0; i < 2; ++i) {
      const _Float16* p = &As[buf][(mw * 32 + i * 16 + lo16) * 32 + hi * 16];
      a[i] = cat8(*(const v8h*)p, *(const v8h*)(p + 8));
    }
#pragma unroll
    for (int j = 0; j < 2; ++j) {
      const _Float16* p = &Bs[buf][(nw * 32 + j * 16 + lo16) * 32 + hi * 16];
      b[j] = *(const v16h*)p;
    }
#pragma unroll
    for (int i = 0; i < 2; ++i)
#pragma unroll
      for (int j = 0; j < 2; ++j) c[i][j] = wmma_f16(a[i], b[j], c[i][j]);
    __syncthreads();
  }

  // epilogue: C layout lane l -> N=l&15, VGPR r -> M=r+8*(l>>4)
#pragma unroll
  for (int j = 0; j < 2; ++j) {
    int n = nBase + nw * 32 + j * 16 + lo16;
    float bv = bias[n];
#pragma unroll
    for (int i = 0; i < 2; ++i) {
      int m0 = mBase + mw * 32 + i * 16 + hi * 8;
      if (MODE == 1) {
        // transposed: per-lane contiguous along m -> one packed 16B store
        v8h pk;
#pragma unroll
        for (int r = 0; r < 8; ++r) pk[r] = (_Float16)(c[i][j][r] + bv);
        *(v8h*)((_Float16*)outp + (size_t)n * M + m0) = pk;
      } else {
#pragma unroll
        for (int r = 0; r < 8; ++r) {
          float val = c[i][j][r] + bv;
          int m = m0 + r;
          if (MODE == 0)
            ((_Float16*)outp)[(size_t)m * N + n] = (_Float16)val;
          else
            ((float*)outp)[(size_t)m * N + n] = val;
        }
      }
    }
  }
}

// ---------------------------------------------------------------------------
// Flash attention: one wave per (head, 16-query block); KV step 64.
// Softmax in exp2 domain: scores scaled by (1/sqrt(Dh))*log2(e).
// ---------------------------------------------------------------------------
__global__ __launch_bounds__(32) void flash_attn(
    const _Float16* __restrict__ qp, const _Float16* __restrict__ kp,
    const _Float16* __restrict__ vT, _Float16* __restrict__ attn) {
  const int lane = threadIdx.x;
  const int lo16 = lane & 15;
  const int hi = lane >> 4;
  const int qb = blockIdx.x;       // 0..255
  const int head = blockIdx.y;     // 0..15
  const int hcol = head * D_HEAD;

  __shared__ __align__(64) _Float16 pbuf[16 * 64];

  const float SCALE = 0.125f * 1.44269504088896340736f;  // 1/sqrt(64)*log2(e)

  // Q fragments (16 rows x 64 K) = two 16x32 A-fragments, loaded once
  v16h qf[2];
#pragma unroll
  for (int c = 0; c < 2; ++c) {
    const _Float16* p =
        qp + (size_t)(qb * 16 + lo16) * D_MODEL + hcol + c * 32 + hi * 16;
    qf[c] = cat8(*(const v8h*)p, *(const v8h*)(p + 8));
  }

  v8f o[4] = {};
  float mrow[8], lrow[8];
#pragma unroll
  for (int j = 0; j < 8; ++j) { mrow[j] = -1e30f; lrow[j] = 0.0f; }

  for (int kb = 0; kb < L_SEQ; kb += 64) {
    // prefetch next KV block (64 K rows; 64 V^T rows) into caches
    if (kb + 64 < L_SEQ) {
      const _Float16* pk = kp + (size_t)(kb + 64 + lane) * D_MODEL + hcol;
      __builtin_prefetch(pk, 0, 3);
      __builtin_prefetch(pk + (size_t)32 * D_MODEL, 0, 3);
      const _Float16* pv = vT + (size_t)(hcol + lane) * L_SEQ + kb + 64;
      __builtin_prefetch(pv, 0, 3);
      __builtin_prefetch(pv + (size_t)32 * L_SEQ, 0, 3);
    }

    // scores: four 16x16 chunks over KV columns kb..kb+63
    v8f s[4];
#pragma unroll
    for (int ch = 0; ch < 4; ++ch) {
      int kvn = kb + ch * 16 + lo16;  // this lane's KV row (B column)
      const _Float16* kr = kp + (size_t)kvn * D_MODEL + hcol;
      v16h kf0 = *(const v16h*)(kr + hi * 16);
      v16h kf1 = *(const v16h*)(kr + 32 + hi * 16);
      v8f acc = {};
      acc = wmma_f16(qf[0], kf0, acc);
      acc = wmma_f16(qf[1], kf1, acc);
      s[ch] = acc * SCALE;
    }

    // streaming softmax stats (rows j within this lane's hi group)
    float alpha[8];
#pragma unroll
    for (int j = 0; j < 8; ++j) {
      float cur = fmaxf(fmaxf(s[0][j], s[1][j]), fmaxf(s[2][j], s[3][j]));
      cur = redmax16(cur);
      float mn = fmaxf(mrow[j], cur);
      alpha[j] = exp2f(mrow[j] - mn);
      mrow[j] = mn;
    }
#pragma unroll
    for (int ch = 0; ch < 4; ++ch)
#pragma unroll
      for (int j = 0; j < 8; ++j) s[ch][j] = exp2f(s[ch][j] - mrow[j]);
#pragma unroll
    for (int j = 0; j < 8; ++j) {
      float rs = redsum16(s[0][j] + s[1][j] + s[2][j] + s[3][j]);
      lrow[j] = lrow[j] * alpha[j] + rs;
    }
#pragma unroll
    for (int t = 0; t < 4; ++t)
#pragma unroll
      for (int j = 0; j < 8; ++j) o[t][j] *= alpha[j];

    // C-layout f32 P -> LDS f16 -> reload as two 16x32 A fragments
    __syncthreads();
#pragma unroll
    for (int ch = 0; ch < 4; ++ch)
#pragma unroll
      for (int j = 0; j < 8; ++j)
        pbuf[(j + 8 * hi) * 64 + ch * 16 + lo16] = (_Float16)s[ch][j];
    __syncthreads();
    v16h pf[2];
#pragma unroll
    for (int f = 0; f < 2; ++f) {
      const _Float16* pp = pbuf + lo16 * 64 + f * 32 + hi * 16;
      pf[f] = cat8(*(const v8h*)pp, *(const v8h*)(pp + 8));
    }

    // O += P(16x64) x V(64x64); V^T gives contiguous KV per lane column
#pragma unroll
    for (int t = 0; t < 4; ++t) {
      const _Float16* vr = vT + (size_t)(hcol + t * 16 + lo16) * L_SEQ + kb;
#pragma unroll
      for (int f = 0; f < 2; ++f) {
        v16h vf = *(const v16h*)(vr + f * 32 + hi * 16);
        o[t] = wmma_f16(pf[f], vf, o[t]);
      }
    }
  }

  // normalize and store f16 (L,D)
#pragma unroll
  for (int t = 0; t < 4; ++t)
#pragma unroll
    for (int j = 0; j < 8; ++j) {
      float val = o[t][j] / lrow[j];
      int m = qb * 16 + j + 8 * hi;
      attn[(size_t)m * D_MODEL + hcol + t * 16 + lo16] = (_Float16)val;
    }
}

// ---------------------------------------------------------------------------
// host-side launcher
// ---------------------------------------------------------------------------
extern "C" void kernel_launch(void* const* d_in, const int* in_sizes, int n_in,
                              void* d_out, int out_size, void* d_ws,
                              size_t ws_size, hipStream_t stream) {
  const float* Q = (const float*)d_in[0];
  const float* K = (const float*)d_in[1];
  const float* V = (const float*)d_in[2];
  const float* w_q = (const float*)d_in[3];
  const float* b_q = (const float*)d_in[4];
  const float* w_k = (const float*)d_in[5];
  const float* b_k = (const float*)d_in[6];
  const float* w_v = (const float*)d_in[7];
  const float* b_v = (const float*)d_in[8];
  const float* w_o = (const float*)d_in[9];
  const float* b_o = (const float*)d_in[10];

  const size_t LD = (size_t)L_SEQ * D_MODEL;
  const size_t DD = (size_t)D_MODEL * D_MODEL;
  const size_t MB = 1u << 20;

  char* ws = (char*)d_ws;
  _Float16* Qh = (_Float16*)(ws + 0 * MB);
  _Float16* Kh = (_Float16*)(ws + 8 * MB);
  _Float16* Vh = (_Float16*)(ws + 16 * MB);
  _Float16* Wq = (_Float16*)(ws + 24 * MB);
  _Float16* Wk = (_Float16*)(ws + 26 * MB);
  _Float16* Wv = (_Float16*)(ws + 28 * MB);
  _Float16* Wo = (_Float16*)(ws + 30 * MB);
  _Float16* qp = (_Float16*)(ws + 32 * MB);
  _Float16* kp = (_Float16*)(ws + 40 * MB);
  _Float16* vT = (_Float16*)(ws + 48 * MB);
  _Float16* at = (_Float16*)(ws + 56 * MB);  // total 64 MB

  {
    int n4 = (int)(LD / 4);
    int blocks = (n4 + 255) / 256;
    cvt_f32_f16<<<blocks, 256, 0, stream>>>(Q, Qh, n4);
    cvt_f32_f16<<<blocks, 256, 0, stream>>>(K, Kh, n4);
    cvt_f32_f16<<<blocks, 256, 0, stream>>>(V, Vh, n4);
    int m4 = (int)(DD / 4);
    int mblocks = (m4 + 255) / 256;
    cvt_f32_f16<<<mblocks, 256, 0, stream>>>(w_q, Wq, m4);
    cvt_f32_f16<<<mblocks, 256, 0, stream>>>(w_k, Wk, m4);
    cvt_f32_f16<<<mblocks, 256, 0, stream>>>(w_v, Wv, m4);
    cvt_f32_f16<<<mblocks, 256, 0, stream>>>(w_o, Wo, m4);
  }

  dim3 ggrid(D_MODEL / 64, L_SEQ / 128);
  gemm_bt<0><<<ggrid, 256, 0, stream>>>(Qh, Wq, b_q, (void*)qp, L_SEQ, D_MODEL,
                                        D_MODEL);
  gemm_bt<0><<<ggrid, 256, 0, stream>>>(Kh, Wk, b_k, (void*)kp, L_SEQ, D_MODEL,
                                        D_MODEL);
  gemm_bt<1><<<ggrid, 256, 0, stream>>>(Vh, Wv, b_v, (void*)vT, L_SEQ, D_MODEL,
                                        D_MODEL);

  flash_attn<<<dim3(L_SEQ / 16, N_HEADS), 32, 0, stream>>>(qp, kp, vT, at);

  gemm_bt<2><<<ggrid, 256, 0, stream>>>(at, Wo, b_o, d_out, L_SEQ, D_MODEL,
                                        D_MODEL);
}
